// LSTM_4423816315549
// MI455X (gfx1250) — matrix operand order repaired and checked by reading, not compiled, not run on hardware
//
#include <hip/hip_runtime.h>
#include <hip/hip_bf16.h>

// ---------------------------------------------------------------------------
// 2-layer LSTM for MI455X (gfx1250): persistent per-layer kernel, bf16 WMMA
// 16x16x32 with fp32 accumulation, fp32 state carry, grid-wide barrier/step.
// ---------------------------------------------------------------------------

typedef __bf16 bf16;
typedef __attribute__((ext_vector_type(16))) __bf16 v16bf;
typedef __attribute__((ext_vector_type(8)))  float  v8f;

constexpr int B_   = 32;
constexpr int T_   = 512;
constexpr int DIN  = 512;
constexpr int H_   = 1024;
constexpr int G4H  = 4 * H_;
constexpr int NWG  = 16;                 // workgroups in recurrent kernel
constexpr int CW   = H_ / NWG;           // 64 h-columns per workgroup
static_assert(CW == 64, "tiling assumes 64 cols/WG");

// ---------------------------------------------------------------------------
// Fragment loaders (bf16 16x16x32 WMMA layouts, wave32)
// A (16xK tile of row-major [rows][ld]):
//   lanes 0-15  : row m0+l, VGPR0-3 = K k0..k0+7,  VGPR4-7 = K k0+16..k0+23
//   lanes 16-31 : row m0+l, VGPR0-3 = K k0+8..15,  VGPR4-7 = K k0+24..31
// B (Kx16 tile; column n of B = row n of row-major weight W[N][ld]):
//   lanes 0-15  : col n0+l, K k0..k0+7, k0+16..k0+23 ; lanes 16-31 : K k0+8..15, k0+24..31
// ---------------------------------------------------------------------------
__device__ __forceinline__ v16bf load_frag_A(const bf16* base, int ld, int m0,
                                             int k0, int lane) {
    const int l = lane & 15, hh = lane >> 4;
    const bf16* p = base + (size_t)(m0 + l) * ld + k0 + hh * 8;
    union { uint4 u[2]; v16bf v; } r;
    r.u[0] = *reinterpret_cast<const uint4*>(p);
    r.u[1] = *reinterpret_cast<const uint4*>(p + 16);
    return r.v;
}

__device__ __forceinline__ v16bf load_frag_B(const bf16* W, int ld, int n0,
                                             int k0, int lane) {
    const int l = lane & 15, hh = lane >> 4;
    const bf16* p = W + (size_t)(n0 + l) * ld + k0 + hh * 8;
    union { uint4 u[2]; v16bf v; } r;
    r.u[0] = *reinterpret_cast<const uint4*>(p);
    r.u[1] = *reinterpret_cast<const uint4*>(p + 16);
    return r.v;
}

__device__ __forceinline__ v8f wmma_bf16(v16bf a, v16bf b, v8f c) {
    return __builtin_amdgcn_wmma_f32_16x16x32_bf16(
        /*neg_a=*/false, a, /*neg_b=*/false, b,
        /*c_mod=*/(short)0, c, /*reuse_a=*/false, /*reuse_b=*/false);
}

// ---------------------------------------------------------------------------
// Sense-reversing grid barrier (NWG workgroups, all resident)
// ---------------------------------------------------------------------------
__device__ __forceinline__ void grid_sync(unsigned* cnt, unsigned* gen) {
    __syncthreads();
    if (threadIdx.x == 0) {
        __threadfence();
        unsigned my = atomicAdd(gen, 0u);
        unsigned prev = atomicAdd(cnt, 1u);
        if (prev == NWG - 1) {
            atomicExch(cnt, 0u);
            __threadfence();
            atomicAdd(gen, 1u);
        } else {
            while (atomicAdd(gen, 0u) == my) {
                __builtin_amdgcn_s_sleep(8);
            }
        }
    }
    __syncthreads();
}

__device__ __forceinline__ float sigmoidf_(float x) {
    return 1.0f / (1.0f + expf(-x));
}

// ---------------------------------------------------------------------------
// Prep kernels
// ---------------------------------------------------------------------------
__global__ void k_transpose_x(const float* __restrict__ x, bf16* __restrict__ xt) {
    // x: [B, DIN, T] -> xt: [T, B, DIN] (bf16)
    size_t idx = (size_t)blockIdx.x * blockDim.x + threadIdx.x;
    size_t n = (size_t)T_ * B_ * DIN;
    if (idx >= n) return;
    int t = (int)(idx / (B_ * DIN));
    int r = (int)(idx % (B_ * DIN));
    int b = r / DIN;
    int d = r % DIN;
    xt[idx] = (bf16)x[((size_t)b * DIN + d) * T_ + t];
}

__global__ void k_cvt_bf16(const float* __restrict__ src, bf16* __restrict__ dst,
                           size_t n) {
    size_t idx = (size_t)blockIdx.x * blockDim.x + threadIdx.x;
    if (idx < n) dst[idx] = (bf16)src[idx];
}

// ---------------------------------------------------------------------------
// Persistent LSTM layer kernel. Grid = NWG blocks x 256 threads (8 waves).
// Wave w: gate g = w>>1, column sub-tile nb = (w&1)*32 within this WG's 64
// h-columns. 4 C-tiles (2 M x 2 N) per wave, K-loop over DIN then H.
// ---------------------------------------------------------------------------
__global__ __launch_bounds__(256)
void lstm_layer_kernel(const bf16* __restrict__ xA, int Kx,
                       const bf16* __restrict__ Wi,   // [4H][Kx] bf16
                       const bf16* __restrict__ Wh,   // [4H][H]  bf16
                       const float* __restrict__ bias,
                       const int*   __restrict__ length,
                       const float* __restrict__ h0,  // [B][H] layer slice
                       const float* __restrict__ c0,
                       float* hstate, float* cstate,  // [B][H] fp32 masters
                       bf16*  hbuf,                   // 2 x [B][H] ping-pong
                       float* out_f32,                // [T][B][H] or null
                       bf16*  out_bf16,               // [T][B][H] or null
                       float* hT_out, float* cT_out,  // [B][H] final state
                       unsigned* bar_cnt, unsigned* bar_gen) {
    __shared__ float gLDS[4][B_][CW];    // 32 KB gate exchange

    const int tid  = threadIdx.x;
    const int wave = tid >> 5;
    const int lane = tid & 31;
    const int c0w  = blockIdx.x * CW;    // first h-column owned by this WG
    const int g    = wave >> 1;          // gate 0..3 (i,f,g,o)
    const int nb   = (wave & 1) * 32;    // column sub-tile within WG
    const int hh   = lane >> 4;
    const int l15  = lane & 15;

    // -------- initialize state slice (each WG owns its 64 columns) --------
    for (int i = tid; i < B_ * CW; i += 256) {
        const int m = i >> 6, nl = i & 63;
        const int gidx = m * H_ + c0w + nl;
        const float hv = h0[gidx], cv = c0[gidx];
        hstate[gidx] = hv;
        cstate[gidx] = cv;
        hbuf[gidx] = (bf16)hv;           // parity-0 buffer for t = 0
    }
    grid_sync(bar_cnt, bar_gen);

    const int n0 = g * H_ + c0w + nb;    // first gate-column row of W

    for (int t = 0; t < T_; ++t) {
        const bf16* hb_read  = hbuf + (size_t)(t & 1) * (B_ * H_);
        bf16*       hb_write = hbuf + (size_t)((t + 1) & 1) * (B_ * H_);

        v8f acc00 = {}, acc01 = {}, acc10 = {}, acc11 = {};

        // ---- x_t @ Wi^T contribution -------------------------------------
        {
            const bf16* xrow = xA + (size_t)t * B_ * Kx;
            for (int k = 0; k < Kx; k += 32) {
                v16bf a0 = load_frag_A(xrow, Kx, 0,  k, lane);
                v16bf a1 = load_frag_A(xrow, Kx, 16, k, lane);
                v16bf b0 = load_frag_B(Wi, Kx, n0,      k, lane);
                v16bf b1 = load_frag_B(Wi, Kx, n0 + 16, k, lane);
                acc00 = wmma_bf16(a0, b0, acc00);
                acc01 = wmma_bf16(a0, b1, acc01);
                acc10 = wmma_bf16(a1, b0, acc10);
                acc11 = wmma_bf16(a1, b1, acc11);
            }
        }
        // ---- h_{t-1} @ Wh^T contribution ---------------------------------
        {
            for (int k = 0; k < H_; k += 32) {
                v16bf a0 = load_frag_A(hb_read, H_, 0,  k, lane);
                v16bf a1 = load_frag_A(hb_read, H_, 16, k, lane);
                v16bf b0 = load_frag_B(Wh, H_, n0,      k, lane);
                v16bf b1 = load_frag_B(Wh, H_, n0 + 16, k, lane);
                acc00 = wmma_bf16(a0, b0, acc00);
                acc01 = wmma_bf16(a0, b1, acc01);
                acc10 = wmma_bf16(a1, b0, acc10);
                acc11 = wmma_bf16(a1, b1, acc11);
            }
        }

        // ---- scatter C tiles to LDS gate buffer --------------------------
        // VGPR r: lanes 0-15 -> row mt*16 + r,     col nb + nt*16 + l
        //         lanes 16-31-> row mt*16 + 8 + r, col nb + nt*16 + l
        {
            const int rb = hh * 8;
#pragma unroll
            for (int r = 0; r < 8; ++r) {
                gLDS[g][ 0 + rb + r][nb +  0 + l15] = acc00[r];
                gLDS[g][ 0 + rb + r][nb + 16 + l15] = acc01[r];
                gLDS[g][16 + rb + r][nb +  0 + l15] = acc10[r];
                gLDS[g][16 + rb + r][nb + 16 + l15] = acc11[r];
            }
        }
        __syncthreads();

        // ---- elementwise gate math, masking, state update ----------------
        for (int i = tid; i < B_ * CW; i += 256) {
            const int m = i >> 6, nl = i & 63;
            const int hcol = c0w + nl;
            const int gidx = m * H_ + hcol;

            const float xi = gLDS[0][m][nl] + bias[0 * H_ + hcol];
            const float xf = gLDS[1][m][nl] + bias[1 * H_ + hcol];
            const float xg = gLDS[2][m][nl] + bias[2 * H_ + hcol];
            const float xo = gLDS[3][m][nl] + bias[3 * H_ + hcol];

            const float ig = sigmoidf_(xi);
            const float fg = sigmoidf_(xf);
            const float gg = tanhf(xg);
            const float og = sigmoidf_(xo);

            const float cp = cstate[gidx];
            const float hp = hstate[gidx];
            const float cn = fg * cp + ig * gg;
            const float hn = og * tanhf(cn);

            const bool live = t < length[m];
            const float hN = live ? hn : hp;
            const float cN = live ? cn : cp;

            hstate[gidx] = hN;
            cstate[gidx] = cN;
            hb_write[gidx] = (bf16)hN;

            const size_t oidx = ((size_t)t * B_ + m) * H_ + hcol;
            if (out_f32)  out_f32[oidx]  = hN;
            if (out_bf16) out_bf16[oidx] = (bf16)hN;
            if (t == T_ - 1) { hT_out[gidx] = hN; cT_out[gidx] = cN; }
        }

        grid_sync(bar_cnt, bar_gen);     // h ping-pong visible to all WGs
    }
}

// ---------------------------------------------------------------------------
// Workspace layout (bytes, 256-aligned)
// ---------------------------------------------------------------------------
constexpr size_t XT_OFF   = 0;                                   // 16 MiB
constexpr size_t WI0_OFF  = XT_OFF  + (size_t)T_ * B_ * DIN * 2; // 4 MiB
constexpr size_t WH0_OFF  = WI0_OFF + (size_t)G4H * DIN * 2;     // 8 MiB
constexpr size_t WI1_OFF  = WH0_OFF + (size_t)G4H * H_ * 2;      // 8 MiB
constexpr size_t WH1_OFF  = WI1_OFF + (size_t)G4H * H_ * 2;      // 8 MiB
constexpr size_t OUT0_OFF = WH1_OFF + (size_t)G4H * H_ * 2;      // 32 MiB
constexpr size_t HB0_OFF  = OUT0_OFF + (size_t)T_ * B_ * H_ * 2; // 2x64 KiB
constexpr size_t HB1_OFF  = HB0_OFF + (size_t)2 * B_ * H_ * 2;
constexpr size_t HS0_OFF  = HB1_OFF + (size_t)2 * B_ * H_ * 2;
constexpr size_t HS1_OFF  = HS0_OFF + (size_t)B_ * H_ * 4;
constexpr size_t CS0_OFF  = HS1_OFF + (size_t)B_ * H_ * 4;
constexpr size_t CS1_OFF  = CS0_OFF + (size_t)B_ * H_ * 4;
constexpr size_t BAR_OFF  = CS1_OFF + (size_t)B_ * H_ * 4;       // 4 uints

extern "C" void kernel_launch(void* const* d_in, const int* in_sizes, int n_in,
                              void* d_out, int out_size, void* d_ws, size_t ws_size,
                              hipStream_t stream) {
    const float* x      = (const float*)d_in[0];
    const int*   length = (const int*)  d_in[1];
    const float* h0     = (const float*)d_in[2];   // [2][B][H]
    const float* c0     = (const float*)d_in[3];
    const float* Wi0    = (const float*)d_in[4];
    const float* Wh0    = (const float*)d_in[5];
    const float* b0     = (const float*)d_in[6];
    const float* Wi1    = (const float*)d_in[7];
    const float* Wh1    = (const float*)d_in[8];
    const float* b1     = (const float*)d_in[9];

    char* ws = (char*)d_ws;
    bf16* xt      = (bf16*)(ws + XT_OFF);
    bf16* wi0b    = (bf16*)(ws + WI0_OFF);
    bf16* wh0b    = (bf16*)(ws + WH0_OFF);
    bf16* wi1b    = (bf16*)(ws + WI1_OFF);
    bf16* wh1b    = (bf16*)(ws + WH1_OFF);
    bf16* out0b   = (bf16*)(ws + OUT0_OFF);
    bf16* hb0     = (bf16*)(ws + HB0_OFF);
    bf16* hb1     = (bf16*)(ws + HB1_OFF);
    float* hs0    = (float*)(ws + HS0_OFF);
    float* hs1    = (float*)(ws + HS1_OFF);
    float* cs0    = (float*)(ws + CS0_OFF);
    float* cs1    = (float*)(ws + CS1_OFF);
    unsigned* bar = (unsigned*)(ws + BAR_OFF);

    float* out1 = (float*)d_out;                       // [T][B][H]
    float* hN   = out1 + (size_t)T_ * B_ * H_;         // [2][B][H]
    float* cN   = hN + (size_t)2 * B_ * H_;            // [2][B][H]

    hipMemsetAsync(bar, 0, 4 * sizeof(unsigned), stream);

    // ---- prep: transpose x, convert weights to bf16 ----------------------
    {
        size_t n = (size_t)T_ * B_ * DIN;
        k_transpose_x<<<(unsigned)((n + 255) / 256), 256, 0, stream>>>(x, xt);
    }
    {
        size_t n = (size_t)G4H * DIN;
        k_cvt_bf16<<<(unsigned)((n + 255) / 256), 256, 0, stream>>>(Wi0, wi0b, n);
    }
    {
        size_t n = (size_t)G4H * H_;
        k_cvt_bf16<<<(unsigned)((n + 255) / 256), 256, 0, stream>>>(Wh0, wh0b, n);
        k_cvt_bf16<<<(unsigned)((n + 255) / 256), 256, 0, stream>>>(Wi1, wi1b, n);
        k_cvt_bf16<<<(unsigned)((n + 255) / 256), 256, 0, stream>>>(Wh1, wh1b, n);
    }

    // ---- layer 0: x(bf16, Din=512) -> out0 (bf16 to ws) ------------------
    lstm_layer_kernel<<<NWG, 256, 0, stream>>>(
        xt, DIN, wi0b, wh0b, b0, length,
        h0 + 0 * B_ * H_, c0 + 0 * B_ * H_,
        hs0, cs0, hb0,
        /*out_f32=*/nullptr, /*out_bf16=*/out0b,
        hN + 0 * B_ * H_, cN + 0 * B_ * H_,
        bar + 0, bar + 1);

    // ---- layer 1: out0(bf16, K=1024) -> out1 (fp32 to d_out) -------------
    lstm_layer_kernel<<<NWG, 256, 0, stream>>>(
        out0b, H_, wi1b, wh1b, b1, length,
        h0 + 1 * B_ * H_, c0 + 1 * B_ * H_,
        hs1, cs1, hb1,
        /*out_f32=*/out1, /*out_bf16=*/nullptr,
        hN + 1 * B_ * H_, cN + 1 * B_ * H_,
        bar + 2, bar + 3);
}
